// SWE_Pooling_59502476918984
// MI455X (gfx1250) — compile-verified, matrix operand order checked by reading, not run on hardware
//
#include <hip/hip_runtime.h>
#include <hip/hip_bf16.h>

// Problem constants (match reference)
#define B_ 256
#define N_ 1024
#define D_ 64
#define L_ 128
#define M_ 128
#define EPS_VAL 1e-3f
#define INTERP_EPS 1e-12f
#define BIG_ (-1e10f)

typedef float    v2f  __attribute__((ext_vector_type(2)));
typedef float    v8f  __attribute__((ext_vector_type(8)));
typedef _Float16 v16h __attribute__((ext_vector_type(16)));

// ---------------------------------------------------------------------------
// 1) W = theta_v / ||theta_v||_row   (one wave32 per row, D=64 -> 2 elems/lane)
// ---------------------------------------------------------------------------
__global__ void normalize_w_kernel(const float* __restrict__ theta,
                                   float* __restrict__ W) {
    int l = blockIdx.x;          // 0..L-1
    int t = threadIdx.x;         // 0..31 (one wave)
    float v0 = theta[l * D_ + t];
    float v1 = theta[l * D_ + 32 + t];
    float ss = v0 * v0 + v1 * v1;
    #pragma unroll
    for (int off = 16; off > 0; off >>= 1)
        ss += __shfl_xor(ss, off, 32);
    float n = sqrtf(ss);
    W[l * D_ + t]      = v0 / n;
    W[l * D_ + 32 + t] = v1 / n;
}

// ---------------------------------------------------------------------------
// 2) Rsorted = sort(reference_pts, axis=0)  -> [M,L], one block per column
// ---------------------------------------------------------------------------
__global__ void sort_ref_kernel(const float* __restrict__ ref,
                                float* __restrict__ Rs) {
    __shared__ float s[M_];
    int l = blockIdx.x, t = threadIdx.x;   // 128 threads
    s[t] = ref[t * L_ + l];
    __syncthreads();
    for (int k = 2; k <= M_; k <<= 1) {
        for (int j = k >> 1; j > 0; j >>= 1) {
            int p = t ^ j;
            if (p > t) {
                bool up = ((t & k) == 0);
                float a = s[t], c = s[p];
                if ((a > c) == up) { s[t] = c; s[p] = a; }
            }
            __syncthreads();
        }
    }
    Rs[t * L_ + l] = s[t];
}

// ---------------------------------------------------------------------------
// 3) Per-batch precompute (shared across all L slices):
//    cs   = cumsum(~mask)          [B,N]
//    vc   = sum(mask)              [B]
//    xc   = (mask ? x0 : max_x) + EPS*cs,  x0=(n+1)/(1+vc)
// ---------------------------------------------------------------------------
__global__ void precompute_b_kernel(const unsigned char* __restrict__ mask,
                                    float* __restrict__ csv,
                                    float* __restrict__ xcv,
                                    float* __restrict__ vcv) {
    __shared__ float s[N_];
    __shared__ float sm[N_];
    int b = blockIdx.x, t = threadIdx.x;   // 1024 threads
    bool valid = mask[b * N_ + t] != 0;
    s[t] = valid ? 0.f : 1.f;
    __syncthreads();
    // inclusive scan (Hillis-Steele)
    for (int off = 1; off < N_; off <<= 1) {
        float v = (t >= off) ? s[t - off] : 0.f;
        __syncthreads();
        s[t] += v;
        __syncthreads();
    }
    float cs = s[t];
    float vc = (float)N_ - s[N_ - 1];
    float x0 = (float)(t + 1) / (1.f + vc);
    sm[t] = valid ? x0 : BIG_;
    __syncthreads();
    for (int sft = N_ / 2; sft > 0; sft >>= 1) {
        if (t < sft) sm[t] = fmaxf(sm[t], sm[t + sft]);
        __syncthreads();
    }
    float max_x = sm[0];
    csv[b * N_ + t] = cs;
    xcv[b * N_ + t] = (valid ? x0 : max_x) + EPS_VAL * cs;
    if (t == 0) vcv[b] = vc;
}

// ---------------------------------------------------------------------------
// 4) Projection GEMM via WMMA: XslT[b][l][n] = X[b][n][:] . W[l][:]
//    One wave per 16(n) x 16(l) tile; f32 WMMA (K=4) preferred, f16 fallback.
//    Block = 128 threads = 4 waves -> 4 consecutive l-tiles.
// ---------------------------------------------------------------------------
__global__ void proj_gemm_kernel(const float* __restrict__ X,
                                 const float* __restrict__ W,
                                 float* __restrict__ XslT) {
    int b    = blockIdx.x;
    int n0   = blockIdx.y * 16;
    int wave = threadIdx.x >> 5;
    int lane = threadIdx.x & 31;
    int l0   = (blockIdx.z * 4 + wave) * 16;
    int half = lane >> 4;     // lane group 0/1
    int lm   = lane & 15;

    v8f acc = {0.f, 0.f, 0.f, 0.f, 0.f, 0.f, 0.f, 0.f};
    const float* xrow = X + ((size_t)b * N_ + (n0 + lm)) * D_;
    const float* wrow = W + (size_t)(l0 + lm) * D_;

#if __has_builtin(__builtin_amdgcn_wmma_f32_16x16x4_f32)
    // A 16x4 f32: lane(m=lm), VGPR0/1 = K = 2*half + {0,1}; mirrored for B.
    #pragma unroll
    for (int k = 0; k < D_; k += 4) {
        v2f a, bb;
        a.x  = xrow[k + 2 * half];
        a.y  = xrow[k + 2 * half + 1];
        bb.x = wrow[k + 2 * half];
        bb.y = wrow[k + 2 * half + 1];
        acc = __builtin_amdgcn_wmma_f32_16x16x4_f32(
            false, a, false, bb, (short)0, acc, false, false);
    }
#else
    // Fallback: f16 16x16x32 (probe-confirmed builtin), two K-chunks of 32.
    #pragma unroll
    for (int c = 0; c < 2; ++c) {
        v16h a, bb;
        #pragma unroll
        for (int v = 0; v < 8; ++v) {
            int kb = (v < 4 ? 2 * v : 16 + 2 * (v - 4)) + 8 * half + 32 * c;
            a[2 * v]      = (_Float16)xrow[kb];
            a[2 * v + 1]  = (_Float16)xrow[kb + 1];
            bb[2 * v]     = (_Float16)wrow[kb];
            bb[2 * v + 1] = (_Float16)wrow[kb + 1];
        }
        acc = __builtin_amdgcn_wmma_f32_16x16x32_f16(
            false, a, false, bb, (short)0, acc, false, false);
    }
#endif
    // C/D layout: VGPR r, lane -> (M = r + 8*half, Ncol = lm)
    // Store transposed: XslT[b][l0+lm][n0 + 8*half + r] (8 contiguous floats)
    float* orow = XslT + ((size_t)b * L_ + (l0 + lm)) * N_ + n0 + 8 * half;
    #pragma unroll
    for (int r = 0; r < 8; ++r) orow[r] = acc[r];
}

// ---------------------------------------------------------------------------
// 5) Per-(b,l) row: branch logic, bitonic sort (N=1024) in LDS, interpolation.
//    One 256-thread block per (b,l).
// ---------------------------------------------------------------------------
__global__ void swe_row_kernel(const float* __restrict__ XslT,
                               const unsigned char* __restrict__ mask,
                               const float* __restrict__ csv,
                               const float* __restrict__ xcv,
                               const float* __restrict__ vcv,
                               const float* __restrict__ Rs,
                               float* __restrict__ out) {
    __shared__ float s_y[N_];
    __shared__ float s_x[N_];
    __shared__ float s_m1[256];
    __shared__ float s_m2[256];

    int l = blockIdx.x, b = blockIdx.y, t = threadIdx.x;
    float vc = vcv[b];
    float* orow = out + ((size_t)b * L_ + l) * M_;

    if (vc < 0.5f) {                 // zero-valid branch: emb row = 0
        if (t < M_) orow[t] = 0.f;
        return;
    }

    const float* yrow = XslT + ((size_t)b * L_ + l) * N_;
    float yv[4], csr[4];
    bool  invv[4];
    float m1 = BIG_, m2 = BIG_;
    #pragma unroll
    for (int q = 0; q < 4; ++q) {
        int i = t + q * 256;
        float y  = yrow[i];
        bool inv = (mask[b * N_ + i] == 0);
        float ym = inv ? BIG_ : y;
        if (ym > m1) { m2 = m1; m1 = ym; } else if (ym > m2) m2 = ym;
        yv[q] = y; invv[q] = inv;
        csr[q]  = csv[b * N_ + i];
        s_x[i]  = xcv[b * N_ + i];
    }
    s_m1[t] = m1; s_m2[t] = m2;
    __syncthreads();
    for (int s = 128; s > 0; s >>= 1) {
        if (t < s) {
            float a1 = s_m1[t], a2 = s_m2[t];
            float b1 = s_m1[t + s], b2 = s_m2[t + s];
            s_m1[t] = fmaxf(a1, b1);
            s_m2[t] = fmaxf(fminf(a1, b1), fmaxf(a2, b2));
        }
        __syncthreads();
    }
    float mx = s_m1[0], second = s_m2[0];
    bool  multi = (vc > 1.5f);
    float slope = (mx - second) * (1.f + vc);
    float single_val = yrow[0];

    #pragma unroll
    for (int q = 0; q < 4; ++q) {
        int i = t + q * 256;
        float val;
        if (multi) {
            float base = invv[q] ? mx : yv[q];
            val = base + slope * (EPS_VAL * csr[q]);
        } else {
            val = single_val;        // single-valid: broadcast projection[0]
        }
        s_y[i] = val;
    }
    __syncthreads();

    // Bitonic sort of 1024 elements, 256 threads (4 indices each).
    for (int k = 2; k <= N_; k <<= 1) {
        for (int j = k >> 1; j > 0; j >>= 1) {
            #pragma unroll
            for (int q = 0; q < 4; ++q) {
                int i = t + q * 256;
                int p = i ^ j;
                if (p > i) {
                    bool up = ((i & k) == 0);
                    float a = s_y[i], c = s_y[p];
                    if ((a > c) == up) { s_y[i] = c; s_y[p] = a; }
                }
            }
            __syncthreads();
        }
    }

    // Interpolate at M uniform quantiles: xnew[m] = (m+1)/(M+1)
    if (t < M_) {
        float v = (float)(t + 1) / (float)(M_ + 1);
        int lo = 0, hi = N_;
        while (lo < hi) {                       // searchsorted 'left'
            int mid = (lo + hi) >> 1;
            if (s_x[mid] < v) lo = mid + 1; else hi = mid;
        }
        int ind = lo;
        if (ind < 1) ind = 1;
        if (ind > N_ - 1) ind = N_ - 1;
        float xl = s_x[ind - 1], xr = s_x[ind];
        float yl = s_y[ind - 1], yr = s_y[ind];
        float sl = (yr - yl) / (xr - xl + INTERP_EPS);
        float yn = yl + sl * (v - xl);
        orow[t] = Rs[t * L_ + l] - yn;          // monge embedding
    }
}

// ---------------------------------------------------------------------------
// Launch
// ---------------------------------------------------------------------------
extern "C" void kernel_launch(void* const* d_in, const int* in_sizes, int n_in,
                              void* d_out, int out_size, void* d_ws, size_t ws_size,
                              hipStream_t stream) {
    (void)in_sizes; (void)n_in; (void)out_size; (void)ws_size;
    const float*         X     = (const float*)d_in[0];
    const unsigned char* mask  = (const unsigned char*)d_in[1]; // jax bool = 1 byte
    const float*         theta = (const float*)d_in[2];
    const float*         ref   = (const float*)d_in[3];
    float*               out   = (float*)d_out;

    // Workspace carve-up (~130 MB; XslT dominates)
    char*  ws  = (char*)d_ws;
    size_t off = 0;
    auto alloc = [&](size_t bytes) -> char* {
        char* p = ws + off;
        off += (bytes + 255) & ~(size_t)255;
        return p;
    };
    float* W    = (float*)alloc((size_t)L_ * D_ * sizeof(float));
    float* Rs   = (float*)alloc((size_t)M_ * L_ * sizeof(float));
    float* csv  = (float*)alloc((size_t)B_ * N_ * sizeof(float));
    float* xcv  = (float*)alloc((size_t)B_ * N_ * sizeof(float));
    float* vcv  = (float*)alloc((size_t)B_ * sizeof(float));
    float* XslT = (float*)alloc((size_t)B_ * L_ * N_ * sizeof(float));

    hipLaunchKernelGGL(normalize_w_kernel, dim3(L_), dim3(32), 0, stream, theta, W);
    hipLaunchKernelGGL(sort_ref_kernel, dim3(L_), dim3(M_), 0, stream, ref, Rs);
    hipLaunchKernelGGL(precompute_b_kernel, dim3(B_), dim3(N_), 0, stream,
                       mask, csv, xcv, vcv);
    hipLaunchKernelGGL(proj_gemm_kernel, dim3(B_, N_ / 16, L_ / 64), dim3(128), 0,
                       stream, X, W, XslT);
    hipLaunchKernelGGL(swe_row_kernel, dim3(L_, B_), dim3(256), 0, stream,
                       XslT, mask, csv, xcv, vcv, Rs, out);
}
